// HGC_SCN_54932631715893
// MI455X (gfx1250) — matrix operand order repaired and checked by reading, not compiled, not run on hardware
//
#include <hip/hip_runtime.h>
#include <hip/hip_bf16.h>

typedef __attribute__((ext_vector_type(16))) _Float16 v16h;
typedef __attribute__((ext_vector_type(8)))  _Float16 v8h;
typedef __attribute__((ext_vector_type(8)))  float    v8f;
typedef __attribute__((ext_vector_type(4)))  float    v4f;

#define NN 50000
#define MM 1000
#define DD 256
#define EE 800000
#define KP 1024   // padded K for projection GEMM (multiple of 64)

union V16U { v16h v; v8h h[2]; };

// ---------------------------------------------------------------------------
// col_emb (transposed, f16, K-padded): BT[d][k] = (relu(k*w1+b1) @ w2 + b2)[d]
// ---------------------------------------------------------------------------
__global__ __launch_bounds__(256) void hgc_colemb(
    const float* __restrict__ w1, const float* __restrict__ b1,
    const float* __restrict__ w2, const float* __restrict__ b2,
    _Float16* __restrict__ BT)
{
    int k = blockIdx.x * 256 + threadIdx.x;   // 0..1023
    int d = blockIdx.y;                       // 0..255
    float val = 0.f;
    if (k < MM) {
        float kk = (float)k;
        float acc = b2[d];
        #pragma unroll
        for (int j = 0; j < 16; ++j) {
            float h = fmaxf(kk * w1[j] + b1[j], 0.f);
            acc += h * w2[j * DD + d];
        }
        val = acc;
    }
    BT[(size_t)d * KP + k] = (_Float16)val;
}

// ---------------------------------------------------------------------------
// cnt[n] = number of nonzeros in init row n  (one wave per row)
// ---------------------------------------------------------------------------
__global__ __launch_bounds__(256) void hgc_cnt(
    const float* __restrict__ init, float* __restrict__ cnt, int n)
{
    int row  = blockIdx.x * 8 + (threadIdx.x >> 5);
    int lane = threadIdx.x & 31;
    if (row >= n) return;
    const float* p = init + (size_t)row * MM;
    float s = 0.f;
    for (int j = lane; j < MM; j += 32) s += (p[j] != 0.f) ? 1.f : 0.f;
    #pragma unroll
    for (int o = 16; o; o >>= 1) s += __shfl_xor(s, o, 32);
    if (lane == 0) cnt[row] = s;
}

// ---------------------------------------------------------------------------
// Fragment loaders (WMMA 16x16x32 f16 layouts, wave32)
//   A frag (16x32, per lane): two 8-half chunks of one row at K+{lg*16, lg*16+8}
//   B frag (32x16, per lane): 16 contiguous K of one column (BT row-major)
// ---------------------------------------------------------------------------
__device__ __forceinline__ v16h hgc_load_a16(const _Float16* pa /* row base + lg*16 */,
                                             int k)
{
    V16U u;
    u.h[0] = *(const v8h*)(pa + k);
    u.h[1] = *(const v8h*)(pa + k + 8);
    return u.v;
}

__device__ __forceinline__ v16h hgc_load_a_mask(const float* pa /* init row base */,
                                                int k, int lg)
{
    V16U u;
    #pragma unroll
    for (int half = 0; half < 2; ++half) {
        int kk = k + lg * 16 + half * 8;              // multiple of 8
        v8h ch;
        if (kk < MM) {                                // MM%8==0 -> full chunk valid
            v4f x0 = *(const v4f*)(pa + kk);
            v4f x1 = *(const v4f*)(pa + kk + 4);
            #pragma unroll
            for (int t = 0; t < 4; ++t) {
                ch[t]     = (x0[t] != 0.f) ? (_Float16)1.0f : (_Float16)0.0f;
                ch[4 + t] = (x1[t] != 0.f) ? (_Float16)1.0f : (_Float16)0.0f;
            }
        } else {
            #pragma unroll
            for (int t = 0; t < 8; ++t) ch[t] = (_Float16)0.0f;
        }
        u.h[half] = ch;
    }
    return u.v;
}

// ---------------------------------------------------------------------------
// Projection GEMM: emb0 = (mask(init) @ col_emb) / max(cnt,1), 0 where cnt==0
// Block tile 128x128, 8 waves (4x2), wave tile 32x64 (2x4 WMMA tiles).
// Double-buffered K pipeline (K multiple of 64): loads for k+32 issue before
// the WMMAs on k, overlapping VMEM latency with the matrix-core burst.
// ---------------------------------------------------------------------------
__global__ __launch_bounds__(256) void hgc_proj_gemm(
    const float* __restrict__ init, const _Float16* __restrict__ BT,
    const float* __restrict__ cnt, float* __restrict__ C, int Mrows)
{
    const int lane  = threadIdx.x & 31;
    const int wave  = threadIdx.x >> 5;
    const int rowBase = blockIdx.x * 128 + (wave >> 1) * 32;
    const int colBase = blockIdx.y * 128 + (wave & 1) * 64;
    const int ml = lane & 15, lg = lane >> 4;

    const float* pa[2];
    #pragma unroll
    for (int i = 0; i < 2; ++i) {
        int r = rowBase + i * 16 + ml;
        pa[i] = init + (size_t)((r < Mrows) ? r : (Mrows - 1)) * MM;
    }
    const _Float16* pb[4];
    #pragma unroll
    for (int j = 0; j < 4; ++j)
        pb[j] = BT + (size_t)(colBase + j * 16 + ml) * KP + lg * 16;

    v8f acc[2][4] = {};
    v16h a0[2], a1[2], b0[4], b1[4];
    #pragma unroll
    for (int i = 0; i < 2; ++i) a0[i] = hgc_load_a_mask(pa[i], 0, lg);
    #pragma unroll
    for (int j = 0; j < 4; ++j) b0[j] = *(const v16h*)(pb[j]);

    for (int k = 0; k < KP; k += 64) {
        // prefetch k+32, compute on k
        #pragma unroll
        for (int i = 0; i < 2; ++i) a1[i] = hgc_load_a_mask(pa[i], k + 32, lg);
        #pragma unroll
        for (int j = 0; j < 4; ++j) b1[j] = *(const v16h*)(pb[j] + k + 32);
        #pragma unroll
        for (int j = 0; j < 4; ++j)
            #pragma unroll
            for (int i = 0; i < 2; ++i)
                acc[i][j] = __builtin_amdgcn_wmma_f32_16x16x32_f16(
                    false, a0[i], false, b0[j], (short)0, acc[i][j], false, false);
        // prefetch k+64, compute on k+32
        if (k + 64 < KP) {
            #pragma unroll
            for (int i = 0; i < 2; ++i) a0[i] = hgc_load_a_mask(pa[i], k + 64, lg);
            #pragma unroll
            for (int j = 0; j < 4; ++j) b0[j] = *(const v16h*)(pb[j] + k + 64);
        }
        #pragma unroll
        for (int j = 0; j < 4; ++j)
            #pragma unroll
            for (int i = 0; i < 2; ++i)
                acc[i][j] = __builtin_amdgcn_wmma_f32_16x16x32_f16(
                    false, a1[i], false, b1[j], (short)0, acc[i][j], false, false);
    }

    // epilogue: scale by 1/cnt (0 if cnt==0), store
    #pragma unroll
    for (int i = 0; i < 2; ++i) {
        int row0 = rowBase + i * 16 + lg * 8;
        #pragma unroll
        for (int v = 0; v < 8; ++v) {
            int row = row0 + v;
            if (row < Mrows) {
                float cn = cnt[row];
                float s = (cn > 0.5f) ? (1.0f / cn) : 0.0f;
                #pragma unroll
                for (int j = 0; j < 4; ++j)
                    C[(size_t)row * DD + colBase + j * 16 + ml] = s * acc[i][j][v];
            }
        }
    }
}

// ---------------------------------------------------------------------------
// Generic f16 GEMM: C[M][Ncols] = A[M][K] (f16 rowmajor) x BT[Ncols][K] (f16)
// Same tiling + double-buffered K pipeline. K must be a multiple of 64.
// ---------------------------------------------------------------------------
__global__ __launch_bounds__(256) void hgc_gemm_f16(
    const _Float16* __restrict__ A, const _Float16* __restrict__ BT,
    float* __restrict__ C, int Mrows, int K, int Ncols)
{
    const int lane  = threadIdx.x & 31;
    const int wave  = threadIdx.x >> 5;
    const int rowBase = blockIdx.x * 128 + (wave >> 1) * 32;
    const int colBase = blockIdx.y * 128 + (wave & 1) * 64;
    const int ml = lane & 15, lg = lane >> 4;

    const _Float16* pa[2];
    #pragma unroll
    for (int i = 0; i < 2; ++i) {
        int r = rowBase + i * 16 + ml;
        pa[i] = A + (size_t)((r < Mrows) ? r : (Mrows - 1)) * K + lg * 16;
    }
    const _Float16* pb[4];
    #pragma unroll
    for (int j = 0; j < 4; ++j)
        pb[j] = BT + (size_t)(colBase + j * 16 + ml) * K + lg * 16;

    v8f acc[2][4] = {};
    v16h a0[2], a1[2], b0[4], b1[4];
    #pragma unroll
    for (int i = 0; i < 2; ++i) a0[i] = hgc_load_a16(pa[i], 0);
    #pragma unroll
    for (int j = 0; j < 4; ++j) b0[j] = *(const v16h*)(pb[j]);

    for (int k = 0; k < K; k += 64) {
        #pragma unroll
        for (int i = 0; i < 2; ++i) a1[i] = hgc_load_a16(pa[i], k + 32);
        #pragma unroll
        for (int j = 0; j < 4; ++j) b1[j] = *(const v16h*)(pb[j] + k + 32);
        #pragma unroll
        for (int j = 0; j < 4; ++j)
            #pragma unroll
            for (int i = 0; i < 2; ++i)
                acc[i][j] = __builtin_amdgcn_wmma_f32_16x16x32_f16(
                    false, a0[i], false, b0[j], (short)0, acc[i][j], false, false);
        if (k + 64 < K) {
            #pragma unroll
            for (int i = 0; i < 2; ++i) a0[i] = hgc_load_a16(pa[i], k + 64);
            #pragma unroll
            for (int j = 0; j < 4; ++j) b0[j] = *(const v16h*)(pb[j] + k + 64);
        }
        #pragma unroll
        for (int j = 0; j < 4; ++j)
            #pragma unroll
            for (int i = 0; i < 2; ++i)
                acc[i][j] = __builtin_amdgcn_wmma_f32_16x16x32_f16(
                    false, a1[i], false, b1[j], (short)0, acc[i][j], false, false);
    }

    #pragma unroll
    for (int i = 0; i < 2; ++i) {
        int row0 = rowBase + i * 16 + lg * 8;
        #pragma unroll
        for (int v = 0; v < 8; ++v) {
            int row = row0 + v;
            if (row < Mrows) {
                #pragma unroll
                for (int j = 0; j < 4; ++j)
                    C[(size_t)row * Ncols + colBase + j * 16 + ml] = acc[i][j][v];
            }
        }
    }
}

// ---------------------------------------------------------------------------
// small helpers
// ---------------------------------------------------------------------------
__global__ __launch_bounds__(256) void hgc_wcast(    // BT[dout][din] = W[din][dout]
    const float* __restrict__ W, _Float16* __restrict__ BT)
{
    int idx = blockIdx.x * 256 + threadIdx.x;        // 0..65535
    int dout = idx >> 8, din = idx & 255;
    BT[idx] = (_Float16)W[din * DD + dout];
}

__global__ __launch_bounds__(256) void hgc_xcast(
    const float* __restrict__ x, _Float16* __restrict__ xh, int n, int doRelu)
{
    int i = blockIdx.x * 256 + threadIdx.x;
    if (i < n) {
        float v = x[i];
        if (doRelu) v = fmaxf(v, 0.f);
        xh[i] = (_Float16)v;
    }
}

__global__ __launch_bounds__(256) void hgc_deg_init(float* __restrict__ deg, int n)
{
    int i = blockIdx.x * 256 + threadIdx.x;
    if (i < n) deg[i] = 1.0f;     // self-loop weight
}

__global__ __launch_bounds__(256) void hgc_deg_acc(
    const int* __restrict__ col, const float* __restrict__ w,
    float* __restrict__ deg, int e)
{
    int i = blockIdx.x * 256 + threadIdx.x;
    if (i < e) atomicAdd(&deg[col[i]], w[i]);
}

__global__ __launch_bounds__(256) void hgc_dinv(
    const float* __restrict__ deg, float* __restrict__ dinv, int n)
{
    int i = blockIdx.x * 256 + threadIdx.x;
    if (i < n) dinv[i] = rsqrtf(deg[i]);
}

// h[n][d] = dinv[n]^2 * xl[n][d] + b[d]   (self-loop term + bias; also zero-inits)
__global__ __launch_bounds__(256) void hgc_agg_init(
    const float* __restrict__ xl, const float* __restrict__ dinv,
    const float* __restrict__ b, float* __restrict__ h, int n)
{
    int i = blockIdx.x * 256 + threadIdx.x;
    if (i < n) {
        int node = i >> 8, d = i & 255;
        float di = dinv[node];
        h[i] = di * di * xl[i] + b[d];
    }
}

// scatter-add: one wave per edge, 8 floats per lane (256 dims)
__global__ __launch_bounds__(256) void hgc_aggregate(
    const int* __restrict__ row, const int* __restrict__ col,
    const float* __restrict__ w, const float* __restrict__ dinv,
    const float* __restrict__ xl, float* __restrict__ h, int e)
{
    int wid  = (int)((blockIdx.x * 256 + threadIdx.x) >> 5);
    int lane = threadIdx.x & 31;
    if (wid >= e) return;
    int r = row[wid], c = col[wid];
    float norm = dinv[r] * w[wid] * dinv[c];
    const float* src = xl + (size_t)r * DD + lane * 8;
    float*       dst = h  + (size_t)c * DD + lane * 8;
    v4f x0 = *(const v4f*)src;
    v4f x1 = *(const v4f*)(src + 4);
    #pragma unroll
    for (int t = 0; t < 4; ++t) {
        atomicAdd(&dst[t],     norm * x0[t]);
        atomicAdd(&dst[4 + t], norm * x1[t]);
    }
}

// attention over the two paths (relu applied here; h_* hold pre-relu layer-3 out)
__global__ __launch_bounds__(256) void hgc_attn(
    const float* __restrict__ h0, const float* __restrict__ h1,
    const float* __restrict__ att_w, const float* __restrict__ att_b,
    float* __restrict__ out, int n)
{
    int node = (int)((blockIdx.x * 256 + threadIdx.x) >> 5);
    int lane = threadIdx.x & 31;
    if (node >= n) return;
    const float* p0 = h0 + (size_t)node * DD + lane * 8;
    const float* p1 = h1 + (size_t)node * DD + lane * 8;
    float e0[8], e1[8], s0 = 0.f, s1 = 0.f;
    #pragma unroll
    for (int t = 0; t < 8; ++t) {
        float aw = att_w[lane * 8 + t];
        e0[t] = fmaxf(p0[t], 0.f);
        e1[t] = fmaxf(p1[t], 0.f);
        s0 += e0[t] * aw;
        s1 += e1[t] * aw;
    }
    #pragma unroll
    for (int o = 16; o; o >>= 1) {
        s0 += __shfl_xor(s0, o, 32);
        s1 += __shfl_xor(s1, o, 32);
    }
    float ab = att_b[0];
    s0 += ab; s1 += ab;
    float mx = fmaxf(s0, s1);
    float w0 = __expf(s0 - mx), w1 = __expf(s1 - mx);
    float inv = 1.f / (w0 + w1);
    w0 *= inv; w1 *= inv;
    float* po = out + (size_t)node * DD + lane * 8;
    #pragma unroll
    for (int t = 0; t < 8; ++t) po[t] = w0 * e0[t] + w1 * e1[t];
}

// ---------------------------------------------------------------------------
extern "C" void kernel_launch(void* const* d_in, const int* in_sizes, int n_in,
                              void* d_out, int out_size, void* d_ws, size_t ws_size,
                              hipStream_t stream)
{
    (void)in_sizes; (void)n_in; (void)out_size; (void)ws_size;

    const float* init   = (const float*)d_in[0];
    const int*   ei[2]  = { (const int*)d_in[1], (const int*)d_in[3] };
    const float* ea[2]  = { (const float*)d_in[2], (const float*)d_in[4] };
    const float* pw1    = (const float*)d_in[5];
    const float* pb1    = (const float*)d_in[6];
    const float* pw2    = (const float*)d_in[7];
    const float* pb2    = (const float*)d_in[8];
    const float* att_w  = (const float*)d_in[21];
    const float* att_b  = (const float*)d_in[22];
    float* out = (float*)d_out;

    // workspace carving (all buffers written before read)
    char* ws = (char*)d_ws;
    size_t off = 0;
    auto carve = [&](size_t bytes) -> void* {
        void* p = ws + off;
        off = (off + bytes + 255) & ~(size_t)255;
        return p;
    };
    _Float16* colembBT = (_Float16*)carve((size_t)DD * KP * 2);
    _Float16* WhT      = (_Float16*)carve((size_t)DD * DD * 2);
    float* cnt   = (float*)carve((size_t)NN * 4);
    float* deg   = (float*)carve((size_t)NN * 4);
    float* dinv  = (float*)carve((size_t)NN * 4);
    float* emb0  = (float*)carve((size_t)NN * DD * 4);
    _Float16* xh = (_Float16*)carve((size_t)NN * DD * 2);
    float* xl    = (float*)carve((size_t)NN * DD * 4);
    float* hA    = (float*)carve((size_t)NN * DD * 4);
    float* hB    = (float*)carve((size_t)NN * DD * 4);
    float* outP[2];
    outP[0] = (float*)carve((size_t)NN * DD * 4);
    outP[1] = (float*)carve((size_t)NN * DD * 4);

    const int ND = NN * DD;
    dim3 gemmGrid((NN + 127) / 128, DD / 128);   // (391, 2)

    // ---- projection ----
    hgc_colemb<<<dim3(KP / 256, DD), 256, 0, stream>>>(pw1, pb1, pw2, pb2, colembBT);
    hgc_cnt<<<(NN + 7) / 8, 256, 0, stream>>>(init, cnt, NN);
    hgc_proj_gemm<<<gemmGrid, 256, 0, stream>>>(init, colembBT, cnt, emb0, NN);

    // ---- two 3-layer GCN stacks ----
    for (int p = 0; p < 2; ++p) {
        const int* rowp = ei[p];
        const int* colp = ei[p] + EE;

        hgc_deg_init<<<(NN + 255) / 256, 256, 0, stream>>>(deg, NN);
        hgc_deg_acc<<<(EE + 255) / 256, 256, 0, stream>>>(colp, ea[p], deg, EE);
        hgc_dinv<<<(NN + 255) / 256, 256, 0, stream>>>(deg, dinv, NN);

        int base = 9 + p * 6;
        for (int l = 0; l < 3; ++l) {
            const float* W = (const float*)d_in[base + 2 * l];
            const float* b = (const float*)d_in[base + 2 * l + 1];
            const float* xin  = (l == 0) ? emb0 : ((l == 1) ? hA : hB);
            float*       hout = (l == 0) ? hA   : ((l == 1) ? hB : outP[p]);

            hgc_wcast<<<DD * DD / 256, 256, 0, stream>>>(W, WhT);
            hgc_xcast<<<(ND + 255) / 256, 256, 0, stream>>>(xin, xh, ND, l > 0);
            hgc_gemm_f16<<<gemmGrid, 256, 0, stream>>>(xh, WhT, xl, NN, DD, DD);
            hgc_agg_init<<<(ND + 255) / 256, 256, 0, stream>>>(xl, dinv, b, hout, ND);
            hgc_aggregate<<<(EE * 32 + 255) / 256, 256, 0, stream>>>(
                rowp, colp, ea[p], dinv, xl, hout, EE);
        }
    }

    // ---- path attention + final output ----
    hgc_attn<<<(NN * 32 + 255) / 256, 256, 0, stream>>>(
        outP[0], outP[1], att_w, att_b, out, NN);
}
